// PoseEstimationLSTM_18245021073670
// MI455X (gfx1250) — matrix operand order repaired and checked by reading, not compiled, not run on hardware
//
#include <hip/hip_runtime.h>
#include <hip/hip_bf16.h>

typedef __bf16 v16bf __attribute__((ext_vector_type(16)));
typedef float  v8f   __attribute__((ext_vector_type(8)));

#define B_   8
#define T_   16
#define D_   113344
#define H_   512
#define G4_  2048     // 4*H
#define O_   6
#define M0_  128      // B*T
#define KSPLIT 7
#define KC_STEPS 506  // (D_/KSPLIT)/32

union AV { v16bf v; __bf16 e[16]; float4 q[2]; };
union CV { v8f   v; float  e[8];  };

// Split 16 fp32 values into bf16 hi/lo (hi = RN(f), lo = RN(f - hi)).
__device__ inline void split16(const float* f, AV& hi, AV& lo) {
#pragma unroll
  for (int i = 0; i < 16; ++i) {
    __bf16 h = (__bf16)f[i];
    hi.e[i] = h;
    lo.e[i] = (__bf16)(f[i] - (float)h);
  }
}

// 16 contiguous fp32 (unused; W/B tiles use ld2runs, lane offset 8*g)
__device__ inline void ld16(const float* p, float* f) {
  float4 a0 = *(const float4*)(p);
  float4 a1 = *(const float4*)(p + 4);
  float4 a2 = *(const float4*)(p + 8);
  float4 a3 = *(const float4*)(p + 12);
  f[0]=a0.x; f[1]=a0.y; f[2]=a0.z; f[3]=a0.w;  f[4]=a1.x; f[5]=a1.y; f[6]=a1.z; f[7]=a1.w;
  f[8]=a2.x; f[9]=a2.y; f[10]=a2.z; f[11]=a2.w; f[12]=a3.x; f[13]=a3.y; f[14]=a3.z; f[15]=a3.w;
}

// A fp32 pattern: two 8-wide runs at p[0..7] and p[16..23] (p = base + s*32 + 8g)
__device__ inline void ld2runs(const float* p, float* f) {
  float4 a0 = *(const float4*)(p);
  float4 a1 = *(const float4*)(p + 4);
  float4 a2 = *(const float4*)(p + 16);
  float4 a3 = *(const float4*)(p + 20);
  f[0]=a0.x; f[1]=a0.y; f[2]=a0.z; f[3]=a0.w;  f[4]=a1.x; f[5]=a1.y; f[6]=a1.z; f[7]=a1.w;
  f[8]=a2.x; f[9]=a2.y; f[10]=a2.z; f[11]=a2.w; f[12]=a3.x; f[13]=a3.y; f[14]=a3.z; f[15]=a3.w;
}

// A bf16 pattern: two 8-element (16B) runs
__device__ inline AV ldA_bf16(const __bf16* base, int o1) {
  AV a;
  a.q[0] = *(const float4*)(base + o1);
  a.q[1] = *(const float4*)(base + o1 + 16);
  return a;
}

__device__ inline v8f wmma3(const AV& ah, const AV& al, const AV& bh, const AV& bl, v8f acc) {
  acc = __builtin_amdgcn_wmma_f32_16x16x32_bf16(false, ah.v, false, bh.v, (short)0, acc, false, false);
  acc = __builtin_amdgcn_wmma_f32_16x16x32_bf16(false, ah.v, false, bl.v, (short)0, acc, false, false);
  acc = __builtin_amdgcn_wmma_f32_16x16x32_bf16(false, al.v, false, bh.v, (short)0, acc, false, false);
  return acc;
}

// One-time split of x into bf16 hi/lo streams (4 elems/thread, exact grid).
__global__ void __launch_bounds__(256)
split_x(const float* __restrict__ x, __bf16* __restrict__ xh, __bf16* __restrict__ xl)
{
  const size_t i4 = ((size_t)blockIdx.x * 256 + threadIdx.x) * 4;
  float4 v = *(const float4*)(x + i4);
  float f[4] = {v.x, v.y, v.z, v.w};
  union { __bf16 e[4]; uint2 u; } h, l;
#pragma unroll
  for (int i = 0; i < 4; ++i) {
    __bf16 a = (__bf16)f[i];
    h.e[i] = a;
    l.e[i] = (__bf16)(f[i] - (float)a);
  }
  *(uint2*)(xh + i4) = h.u;
  *(uint2*)(xl + i4) = l.u;
}

// Tiled GEMM: out[row,col](+bias) or partial[kc] = A[128,K] @ W[2048,K]^T.
// Block = 256 thr = 8 waves: wave = (mp 0..3)*(2 M-tiles) x (nt 0..1 N-tile).
// Block covers M=128 x N=32; grid.x = 2048/32; grid.y = K-split.
// W converted in-register (cost amortized over 2 M-tiles = 6 WMMAs).
template<bool ASPLIT>
__global__ void __launch_bounds__(256)
gemm_k(const float* __restrict__ Af,
       const __bf16* __restrict__ Ah, const __bf16* __restrict__ Al,
       const float* __restrict__ W, float* __restrict__ out, int K, int ksteps,
       const float* __restrict__ b1, const float* __restrict__ b2)
{
  const int wave = threadIdx.x >> 5;
  const int lane = threadIdx.x & 31;
  const int m = lane & 15;
  const int g = lane >> 4;
  const int mp = wave & 3;     // M-pair: rows [32mp, 32mp+32)
  const int nt = wave >> 2;    // N-tile within block
  const int bx = blockIdx.x;
  const int kc = blockIdx.y;
  const long k0 = (long)kc * ksteps * 32;

  const int colW = bx * 32 + nt * 16 + m;
  const float* Wp = W + (size_t)colW * K + k0 + 8 * g;

  const size_t rowA0 = (size_t)(mp * 32 + m);
  const size_t rowA1 = rowA0 + 16;
  const float*  Af0 = ASPLIT ? nullptr : (Af + rowA0 * K + k0);
  const float*  Af1 = ASPLIT ? nullptr : (Af + rowA1 * K + k0);
  const __bf16* Ah0 = ASPLIT ? (Ah + rowA0 * K + k0) : nullptr;
  const __bf16* Al0 = ASPLIT ? (Al + rowA0 * K + k0) : nullptr;
  const __bf16* Ah1 = ASPLIT ? (Ah + rowA1 * K + k0) : nullptr;
  const __bf16* Al1 = ASPLIT ? (Al + rowA1 * K + k0) : nullptr;

  CV acc0, acc1;
#pragma unroll
  for (int r = 0; r < 8; ++r) { acc0.e[r] = 0.0f; acc1.e[r] = 0.0f; }

  for (int s = 0; s < ksteps; ++s) {
    const float* wp = Wp + s * 32;
    __builtin_prefetch(wp + 256, 0, 1);   // global_prefetch on the W HBM stream
    float wf[16];
    ld2runs(wp, wf);
    AV bh, bl;
    split16(wf, bh, bl);

    AV ah0, al0, ah1, al1;
    if (ASPLIT) {
      const int o1 = s * 32 + 8 * g;
      ah0 = ldA_bf16(Ah0, o1);  al0 = ldA_bf16(Al0, o1);
      ah1 = ldA_bf16(Ah1, o1);  al1 = ldA_bf16(Al1, o1);
    } else {
      float af[16];
      ld2runs(Af0 + s * 32 + 8 * g, af);
      split16(af, ah0, al0);
      ld2runs(Af1 + s * 32 + 8 * g, af);
      split16(af, ah1, al1);
    }
    acc0.v = wmma3(ah0, al0, bh, bl, acc0.v);
    acc1.v = wmma3(ah1, al1, bh, bl, acc1.v);
  }

  float* op = out + (size_t)kc * ((size_t)M0_ * G4_);
  const int col = bx * 32 + nt * 16 + m;
  const float badd = b1 ? (b1[col] + b2[col]) : 0.0f;
#pragma unroll
  for (int r = 0; r < 8; ++r) {
    const int row0 = mp * 32 + r + 8 * g;        // C/D: M = r + 8*(lane>>4)
    op[(size_t)row0 * G4_ + col] = acc0.e[r] + badd;
    op[(size_t)(row0 + 16) * G4_ + col] = acc1.e[r] + badd;
  }
}

// Sum KSPLIT partials + both biases -> xg[128,2048]
__global__ void __launch_bounds__(256)
reduce_bias(const float* __restrict__ part, const float* __restrict__ b1,
            const float* __restrict__ b2, float* __restrict__ xg)
{
  const int idx = blockIdx.x * blockDim.x + threadIdx.x;
  const int col = idx & (G4_ - 1);
  float s = b1[col] + b2[col];
#pragma unroll
  for (int k = 0; k < KSPLIT; ++k) s += part[(size_t)k * ((size_t)M0_ * G4_) + idx];
  xg[idx] = s;
}

// Persistent LSTM recurrence: 64 WGs x 64 thr (2 waves). WG owns 8 hidden units.
// Whh tile pre-split into LDS (bf16 hi/lo, B-lane layout) once.
// A-matrix comes from a zero-padded time-major staging buffer HR[(T+1)][16][512]
// (slot 0 and rows 8..15 are pre-zeroed) -> unconditional loads, no divergence.
__global__ void __launch_bounds__(64)
lstm_recur(const float* __restrict__ xg, const float* __restrict__ Whh,
           float* __restrict__ hseq, __bf16* __restrict__ hhi, __bf16* __restrict__ hlo,
           __bf16* __restrict__ HRh, __bf16* __restrict__ HRl,
           int* __restrict__ bar)
{
  extern __shared__ char smem[];
  __bf16* wh = (__bf16*)smem;             // [2 waves][16 ks][32 lanes][16] hi
  __bf16* wl = wh + 2 * 16 * 32 * 16;     // same, lo
  float*  gl = (float*)(wl + 2 * 16 * 32 * 16);  // [4 gates][8 units][8 batch]

  const int NWG  = 64;
  const int j0   = blockIdx.x * 8;
  const int wave = threadIdx.x >> 5;
  const int lane = threadIdx.x & 31;
  const int n    = lane & 15;
  const int g    = lane >> 4;
  const int gate = wave * 2 + (n >> 3);   // 0:i 1:f 2:g 3:o
  const int unit = j0 + (n & 7);
  const float* wrow = Whh + (size_t)(gate * H_ + unit) * H_;

  // prepack Whh tile -> LDS (one-time)
  for (int s = 0; s < 16; ++s) {
    float wf[16];
    ld2runs(wrow + s * 32 + 8 * g, wf);
    AV bh, bl;
    split16(wf, bh, bl);
    const int idx = ((wave * 16 + s) * 32 + lane) * 16;
    *(float4*)(wh + idx) = bh.q[0];  *(float4*)(wh + idx + 8) = bh.q[1];
    *(float4*)(wl + idx) = bl.q[0];  *(float4*)(wl + idx + 8) = bl.q[1];
  }
  __syncthreads();

  const int u = threadIdx.x >> 3;   // pointwise: unit 0..7
  const int b = threadIdx.x & 7;    // pointwise: batch 0..7
  float c = 0.0f;

  for (int t = 0; t < T_; ++t) {
    CV acc;
#pragma unroll
    for (int r = 0; r < 8; ++r) acc.e[r] = 0.0f;

    // A = h_{t-1}: slot t of HR (slot 0 = zeros; rows 8..15 = zeros)
    const __bf16* hb = HRh + ((size_t)t * 16 + n) * H_;
    const __bf16* lb = HRl + ((size_t)t * 16 + n) * H_;

    for (int s = 0; s < 16; ++s) {
      const int o1 = s * 32 + 8 * g;
      AV ah = ldA_bf16(hb, o1);
      AV al = ldA_bf16(lb, o1);
      AV bh, bl;
      const int idx = ((wave * 16 + s) * 32 + lane) * 16;
      bh.q[0] = *(const float4*)(wh + idx);  bh.q[1] = *(const float4*)(wh + idx + 8);
      bl.q[0] = *(const float4*)(wl + idx);  bl.q[1] = *(const float4*)(wl + idx + 8);
      acc.v = wmma3(ah, al, bh, bl, acc.v);
    }

    if (lane < 16) {            // lanes 0..15 hold batches 0..7 in acc[0..7]
#pragma unroll
      for (int r = 0; r < 8; ++r) gl[(gate * 8 + (n & 7)) * 8 + r] = acc.e[r];
    }
    __syncthreads();

    // pointwise for (unit u, batch b)
    const size_t xrow = (size_t)(b * T_ + t) * G4_;
    const int jc = j0 + u;
    const float xi = xg[xrow + jc]          + gl[(0 * 8 + u) * 8 + b];
    const float xf = xg[xrow + 512  + jc]   + gl[(1 * 8 + u) * 8 + b];
    const float xc = xg[xrow + 1024 + jc]   + gl[(2 * 8 + u) * 8 + b];
    const float xo = xg[xrow + 1536 + jc]   + gl[(3 * 8 + u) * 8 + b];
    const float ig = 1.0f / (1.0f + __expf(-xi));
    const float fg = 1.0f / (1.0f + __expf(-xf));
    const float cg = tanhf(xc);
    const float og = 1.0f / (1.0f + __expf(-xo));
    c = fg * c + ig * cg;
    const float h = og * tanhf(c);

    const size_t hidx = (size_t)(b * T_ + t) * H_ + jc;   // layer-1 GEMM layout
    hseq[hidx] = h;
    __bf16 hh = (__bf16)h;
    __bf16 hl = (__bf16)(h - (float)hh);
    hhi[hidx] = hh;
    hlo[hidx] = hl;
    const size_t ridx = ((size_t)(t + 1) * 16 + b) * H_ + jc;  // next step's A slot
    HRh[ridx] = hh;
    HRl[ridx] = hl;

    __threadfence();
    __syncthreads();
    if (threadIdx.x == 0) {
      atomicAdd(bar, 1);
      while (atomicAdd(bar, 0) < (t + 1) * NWG) __builtin_amdgcn_s_sleep(2);
    }
    __syncthreads();
  }
}

// out[b,o] = h1[b,T-1,:] . fc_w[o,:] + fc_b[o]
__global__ void __launch_bounds__(256)
fc_kernel(const float* __restrict__ hseq, const float* __restrict__ fcw,
          const float* __restrict__ fcb, float* __restrict__ out)
{
  const int bidx = threadIdx.x >> 5;
  const int lane = threadIdx.x & 31;
  const float* h = hseq + (size_t)(bidx * T_ + (T_ - 1)) * H_;
  for (int o = 0; o < O_; ++o) {
    float s = 0.0f;
    for (int k = lane; k < H_; k += 32) s += h[k] * fcw[o * H_ + k];
#pragma unroll
    for (int off = 16; off; off >>= 1) s += __shfl_xor(s, off, 32);
    if (lane == 0) out[bidx * O_ + o] = s + fcb[o];
  }
}

extern "C" void kernel_launch(void* const* d_in, const int* in_sizes, int n_in,
                              void* d_out, int out_size, void* d_ws, size_t ws_size,
                              hipStream_t stream) {
  const float* x    = (const float*)d_in[0];
  const float* Wih0 = (const float*)d_in[1];
  const float* Whh0 = (const float*)d_in[2];
  const float* bih0 = (const float*)d_in[3];
  const float* bhh0 = (const float*)d_in[4];
  const float* Wih1 = (const float*)d_in[5];
  const float* Whh1 = (const float*)d_in[6];
  const float* bih1 = (const float*)d_in[7];
  const float* bhh1 = (const float*)d_in[8];
  const float* fcw  = (const float*)d_in[9];
  const float* fcb  = (const float*)d_in[10];
  float* out = (float*)d_out;

  size_t off = 0;
  auto alloc = [&](size_t bytes) -> char* {
    char* p = (char*)d_ws + off;
    off += (bytes + 255) & ~(size_t)255;
    return p;
  };
  float*  P_part = (float*)alloc((size_t)KSPLIT * M0_ * G4_ * 4);
  float*  P_xg0  = (float*)alloc((size_t)M0_ * G4_ * 4);
  float*  P_xg1  = (float*)alloc((size_t)M0_ * G4_ * 4);
  float*  P_h0   = (float*)alloc((size_t)M0_ * H_ * 4);
  float*  P_h1   = (float*)alloc((size_t)M0_ * H_ * 4);
  __bf16* H0h    = (__bf16*)alloc((size_t)M0_ * H_ * 2);
  __bf16* H0l    = (__bf16*)alloc((size_t)M0_ * H_ * 2);
  __bf16* H1h    = (__bf16*)alloc((size_t)M0_ * H_ * 2);
  __bf16* H1l    = (__bf16*)alloc((size_t)M0_ * H_ * 2);
  const size_t hr_elems = (size_t)(T_ + 1) * 16 * H_;
  __bf16* HRh    = (__bf16*)alloc(hr_elems * 2);
  __bf16* HRl    = (__bf16*)alloc(hr_elems * 2);
  int*    P_bar  = (int*)alloc(256);

  const size_t xsplit_bytes = (size_t)M0_ * D_ * 2;
  const bool presplit = (ws_size >= off + 2 * xsplit_bytes);
  __bf16* Xh = nullptr; __bf16* Xl = nullptr;
  if (presplit) {
    Xh = (__bf16*)alloc(xsplit_bytes);
    Xl = (__bf16*)alloc(xsplit_bytes);
  }

  hipMemsetAsync(P_bar, 0, 2 * sizeof(int), stream);
  hipMemsetAsync(HRh, 0, hr_elems * 2, stream);   // zero pad rows + t=0 slot
  hipMemsetAsync(HRl, 0, hr_elems * 2, stream);

  // ---- Layer 0 input GEMM: W_ih0 streams HBM once; x (L2-resident) pre-split ----
  if (presplit) {
    split_x<<<(M0_ * (D_ / 4)) / 256, 256, 0, stream>>>(x, Xh, Xl);
    gemm_k<true><<<dim3(G4_ / 32, KSPLIT), 256, 0, stream>>>(
        nullptr, Xh, Xl, Wih0, P_part, D_, KC_STEPS, nullptr, nullptr);
  } else {
    gemm_k<false><<<dim3(G4_ / 32, KSPLIT), 256, 0, stream>>>(
        x, nullptr, nullptr, Wih0, P_part, D_, KC_STEPS, nullptr, nullptr);
  }
  reduce_bias<<<(M0_ * G4_) / 256, 256, 0, stream>>>(P_part, bih0, bhh0, P_xg0);

  const int smem = (2 * 16 * 32 * 16) * 2 * 2 + 4 * 8 * 8 * 4;  // 66560 B
  lstm_recur<<<64, 64, smem, stream>>>(P_xg0, Whh0, P_h0, H0h, H0l, HRh, HRl, P_bar + 0);

  // ---- Layer 1 input GEMM on pre-split h0 (bias fused) ----
  gemm_k<true><<<dim3(G4_ / 32, 1), 256, 0, stream>>>(
      nullptr, H0h, H0l, Wih1, P_xg1, H_, H_ / 32, bih1, bhh1);

  lstm_recur<<<64, 64, smem, stream>>>(P_xg1, Whh1, P_h1, H1h, H1l, HRh, HRl, P_bar + 1);

  fc_kernel<<<1, 256, 0, stream>>>(P_h1, fcw, fcb, out);
}
